// MomentumDecoder_74577812128594
// MI455X (gfx1250) — matrix operand order repaired and checked by reading, not compiled, not run on hardware
//
#include <hip/hip_runtime.h>
#include <hip/hip_bf16.h>

// ---------------------------------------------------------------------------
// Problem constants (from reference): B=512, H=512, L=2, PRED_LEN=256
// ---------------------------------------------------------------------------
#define BSZ   512
#define HD    512
#define GD    2048          // 4*H gate dimension
#define TSTEPS 256
#define TB    16            // batch rows per workgroup
#define NWG   (BSZ / TB)    // 32 workgroups
#define WAVES 16            // waves per workgroup (wave32)
#define MATELEMS (GD * HD)  // 2048*512 = 1048576 elements per weight matrix

typedef __attribute__((ext_vector_type(8)))  float   v8f;
typedef __attribute__((ext_vector_type(8)))  __bf16  v8bf;
typedef __attribute__((ext_vector_type(16))) __bf16  v16bf;

// ---------------------------------------------------------------------------
// Activations
// ---------------------------------------------------------------------------
__device__ __forceinline__ float sigmoidf_(float x) {
    return 1.0f / (1.0f + __expf(-x));
}
__device__ __forceinline__ float tanhf_(float x) {
#if __has_builtin(__builtin_amdgcn_tanhf)
    return __builtin_amdgcn_tanhf(x);        // v_tanh_f32 on gfx1250
#else
    float e = __expf(2.0f * x);
    return 1.0f - 2.0f / (e + 1.0f);
#endif
}

// ---------------------------------------------------------------------------
// Prologue: convert weights f32->bf16 packed into per-lane WMMA B-fragments,
// and fuse bias pairs.
//
// B-fragment layout used here for V_WMMA_F32_16X16X32_BF16 (32x16 K-major B; contiguous K, matching A):
//   lane l<16 : column n = nt*16+l,     K = kc*32 + 0..15
//   lane l>=16: column n = nt*16+(l-16),K = kc*32 + 16..31
// Each lane holds 16 contiguous bf16 (32 bytes) -> one fragment = 1024 B.
// Fragment address: wpack[(mat*2048 + nt*16 + kc)*512 + lane*16 + j]
// ---------------------------------------------------------------------------
__global__ void pack_weights_kernel(const float* __restrict__ whh0,
                                    const float* __restrict__ wih1,
                                    const float* __restrict__ whh1,
                                    const float* __restrict__ bih0,
                                    const float* __restrict__ bhh0,
                                    const float* __restrict__ bih1,
                                    const float* __restrict__ bhh1,
                                    __bf16* __restrict__ wpack,
                                    float*  __restrict__ bsum) {
    int tid = blockIdx.x * blockDim.x + threadIdx.x;
    if (tid < 2 * GD) {   // fused biases: bsum[0..2047]=layer0, [2048..4095]=layer1
        bsum[tid] = (tid < GD) ? (bih0[tid] + bhh0[tid])
                               : (bih1[tid - GD] + bhh1[tid - GD]);
    }
    if (tid >= 3 * 2048 * 32) return;          // 3 mats * 2048 frags * 32 lanes
    int lane = tid & 31;
    int frag = (tid >> 5) & 2047;              // nt*16 + kc
    int mat  = tid >> 16;                      // 2048*32 == 1<<16
    int nt = frag >> 4;
    int kc = frag & 15;
    const float* W = (mat == 0) ? whh0 : (mat == 1) ? wih1 : whh1;
    int n     = nt * 16 + (lane & 15);
    int kbase = kc * 32 + (lane >> 4) * 16;
    __bf16* dst = wpack + (size_t)(mat * 2048 + frag) * 512 + lane * 16;
#pragma unroll
    for (int j = 0; j < 16; ++j)
        dst[j] = (__bf16)W[(size_t)n * HD + kbase + j];
}

// ---------------------------------------------------------------------------
// A-fragment: 16x32 bf16 tile of h from row-major LDS.
// A layout used here (contiguous K, matching B): lane l<16 -> row l, K=0..15;
//                  lane l>=16 -> row l-16, K=16..31.
// arow already includes the per-lane-half K sub-offset, so both loads use
// compile-time DS offsets.
// ---------------------------------------------------------------------------
__device__ __forceinline__ v16bf load_a_frag(const __bf16* arow, int kb) {
    v8bf a0 = *(const v8bf*)(arow + kb);
    v8bf a1 = *(const v8bf*)(arow + kb + 8);
    v16bf a;
#pragma unroll
    for (int i = 0; i < 8; ++i) { a[i] = a0[i]; a[8 + i] = a1[i]; }
    return a;
}

// GEMM: acc[g][t] += h(16x512) x W^T block for this wave's 8 n-tiles
// (4 gates x 2 tiles of 16 columns).  128 WMMAs per call.
//
// Addressing: `mat` stays a kernel-arg pointer (SGPR pair); `bofs` is the
// loop-invariant per-wave/per-lane 32-bit element offset; the per-fragment
// part (g,tt,kc all unrolled constants) folds into the 24-bit instruction
// offset -> SADDR-form global_load_b128 with immediate offsets, zero per-
// iteration address math.
// ---------------------------------------------------------------------------
__device__ __forceinline__ void gemm_acc(const __bf16* __restrict__ arow,
                                         const __bf16* __restrict__ mat,
                                         int bofs, v8f (&acc)[4][2]) {
#pragma unroll 4
    for (int kc = 0; kc < 16; ++kc) {
        v16bf a = load_a_frag(arow, kc * 32);
#pragma unroll
        for (int g = 0; g < 4; ++g) {
#pragma unroll
            for (int tt = 0; tt < 2; ++tt) {
                // elements: g*32*16*512 + tt*16*512 + kc*512  (all constants)
                v16bf b = *(const v16bf*)(mat + bofs +
                                          (g * 262144 + tt * 8192 + kc * 512));
                acc[g][tt] = __builtin_amdgcn_wmma_f32_16x16x32_bf16(
                    false, a, false, b, (short)0, acc[g][tt], false, false);
            }
        }
    }
}

// ---------------------------------------------------------------------------
// Main persistent kernel: one WG per 16 batch rows, loops all 256 steps.
// No inter-WG communication needed (recurrence is per-batch-row).
// ---------------------------------------------------------------------------
__global__ __launch_bounds__(512, 1)
void lstm_decode_kernel(const float* __restrict__ hin,   // (L,B,H)
                        const float* __restrict__ cin,   // (L,B,H)
                        const float* __restrict__ wih0,  // (2048,1)
                        const float* __restrict__ fcw,   // (1,512)
                        const float* __restrict__ fcb,   // (1,)
                        const __bf16* __restrict__ wpack0,  // Whh0 packed
                        const __bf16* __restrict__ wpack1,  // Wih1 packed
                        const __bf16* __restrict__ wpack2,  // Whh1 packed
                        const float* __restrict__ bsum,  // fused biases (2*2048)
                        float* __restrict__ out) {       // (B, 256)
    __shared__ __attribute__((aligned(32))) __bf16 h0buf[TB][HD];  // 16 KB
    __shared__ __attribute__((aligned(32))) __bf16 h1buf[TB][HD];  // 16 KB
    __shared__ __attribute__((aligned(32))) float  xbuf[TB];
    __shared__ __attribute__((aligned(32))) float  predacc[TB];

    const int tid  = threadIdx.x;
    const int lane = tid & 31;
    const int wv   = tid >> 5;          // wave 0..15: owns hidden cols [32w,32w+32)
    const int bg   = blockIdx.x * TB;   // batch base row
    const int hi   = lane >> 4;         // lane half
    const int ln   = lane & 15;
    const int koff = hi ? 16 : 0;       // A-fragment K sub-offset per lane half
    const int colbase = wv * 32;

    // loop-invariant 32-bit element offset into packed weights for this thread
    const int bofs = wv * 2 * 8192 + lane * 16;

    // ---- load initial hidden states into LDS (bf16) ----
    for (int idx = tid; idx < TB * HD; idx += 512) {
        int r = idx >> 9, k = idx & (HD - 1);
        h0buf[r][k] = (__bf16)hin[(size_t)(bg + r) * HD + k];
        h1buf[r][k] = (__bf16)hin[(size_t)(BSZ + bg + r) * HD + k];
    }
    if (tid < TB) { xbuf[tid] = 0.0f; predacc[tid] = 0.0f; }

    // ---- per-lane constants: rank-1 W_ih0 column, fused biases, fc weights ----
    float wih0v[4][2], b0v[4][2], b1v[4][2], fcwv[2];
#pragma unroll
    for (int tt = 0; tt < 2; ++tt) {
        int col = colbase + tt * 16 + ln;
        fcwv[tt] = fcw[col];
#pragma unroll
        for (int g = 0; g < 4; ++g) {
            int n = g * HD + col;
            wih0v[g][tt] = wih0[n];
            b0v[g][tt]   = bsum[n];
            b1v[g][tt]   = bsum[GD + n];
        }
    }

    // ---- persistent cell state in registers (C/D WMMA layout) ----
    float c0[2][8], c1[2][8];
#pragma unroll
    for (int tt = 0; tt < 2; ++tt) {
        int col = colbase + tt * 16 + ln;
#pragma unroll
        for (int e = 0; e < 8; ++e) {
            int b = bg + e + 8 * hi;
            c0[tt][e] = cin[(size_t)b * HD + col];
            c1[tt][e] = cin[(size_t)(BSZ + b) * HD + col];
        }
    }

    // hoisted LDS bases: all per-step DS accesses use immediate offsets
    const __bf16* arow0 = &h0buf[ln][0] + koff;  // A-frag source, layer0 h
    const __bf16* arow1 = &h1buf[ln][0] + koff;  // A-frag source, layer1 h
    __bf16* hst0 = &h0buf[8 * hi][colbase + ln]; // elementwise store base
    __bf16* hst1 = &h1buf[8 * hi][colbase + ln];
    float*  pacc = &predacc[8 * hi];
    const float* xsrc = &xbuf[8 * hi];
    const float fcbv = fcb[0];
    __syncthreads();

    for (int t = 0; t < TSTEPS; ++t) {
        // x (previous prediction) for this lane's 8 rows
        float xv[8];
#pragma unroll
        for (int e = 0; e < 8; ++e) xv[e] = xsrc[e];

        // ---------- layer 0: gates = x*Wih0^T + h0*Whh0^T + bias ----------
        v8f acc[4][2];
#pragma unroll
        for (int g = 0; g < 4; ++g)
#pragma unroll
            for (int tt = 0; tt < 2; ++tt)
#pragma unroll
                for (int e = 0; e < 8; ++e)
                    acc[g][tt][e] = xv[e] * wih0v[g][tt] + b0v[g][tt];

        gemm_acc(arow0, wpack0, bofs, acc);
        __syncthreads();                    // all reads of h0buf complete

#pragma unroll
        for (int tt = 0; tt < 2; ++tt) {
#pragma unroll
            for (int e = 0; e < 8; ++e) {
                float iv = sigmoidf_(acc[0][tt][e]);
                float fv = sigmoidf_(acc[1][tt][e]);
                float gv = tanhf_(acc[2][tt][e]);
                float ov = sigmoidf_(acc[3][tt][e]);
                float cn = fv * c0[tt][e] + iv * gv;
                c0[tt][e] = cn;
                hst0[e * HD + tt * 16] = (__bf16)(ov * tanhf_(cn));
            }
        }
        __syncthreads();                    // h0_new visible

        // ---------- layer 1: gates = h0_new*Wih1^T + h1*Whh1^T + bias ----------
#pragma unroll
        for (int g = 0; g < 4; ++g)
#pragma unroll
            for (int tt = 0; tt < 2; ++tt)
#pragma unroll
                for (int e = 0; e < 8; ++e)
                    acc[g][tt][e] = b1v[g][tt];

        gemm_acc(arow0, wpack1, bofs, acc);
        gemm_acc(arow1, wpack2, bofs, acc);
        __syncthreads();                    // all reads of h1buf complete

#pragma unroll
        for (int tt = 0; tt < 2; ++tt) {
#pragma unroll
            for (int e = 0; e < 8; ++e) {
                float iv = sigmoidf_(acc[0][tt][e]);
                float fv = sigmoidf_(acc[1][tt][e]);
                float gv = tanhf_(acc[2][tt][e]);
                float ov = sigmoidf_(acc[3][tt][e]);
                float cn = fv * c1[tt][e] + iv * gv;
                c1[tt][e] = cn;
                float hn = ov * tanhf_(cn);
                hst1[e * HD + tt * 16] = (__bf16)hn;
                atomicAdd(pacc + e, hn * fcwv[tt]);  // ds_add_f32
            }
        }
        __syncthreads();

        if (tid < TB) {
            float p = predacc[tid] + fcbv;
            out[(size_t)(bg + tid) * TSTEPS + t] = p;
            xbuf[tid]   = p;        // next-step input token
            predacc[tid] = 0.0f;
        }
        __syncthreads();
    }
}

// ---------------------------------------------------------------------------
// Launch
// ---------------------------------------------------------------------------
extern "C" void kernel_launch(void* const* d_in, const int* in_sizes, int n_in,
                              void* d_out, int out_size, void* d_ws, size_t ws_size,
                              hipStream_t stream) {
    const float* h    = (const float*)d_in[0];
    const float* c    = (const float*)d_in[1];
    const float* Wih0 = (const float*)d_in[2];
    const float* Whh0 = (const float*)d_in[3];
    const float* bih0 = (const float*)d_in[4];
    const float* bhh0 = (const float*)d_in[5];
    const float* Wih1 = (const float*)d_in[6];
    const float* Whh1 = (const float*)d_in[7];
    const float* bih1 = (const float*)d_in[8];
    const float* bhh1 = (const float*)d_in[9];
    const float* fcw  = (const float*)d_in[10];
    const float* fcb  = (const float*)d_in[11];
    float* out = (float*)d_out;

    // workspace: 3 packed bf16 matrices (6 MB) + fused biases (16 KB)
    __bf16* wpack = (__bf16*)d_ws;
    float*  bsum  = (float*)((char*)d_ws + (size_t)3 * MATELEMS * sizeof(__bf16));

    int pack_threads = 3 * 2048 * 32;
    pack_weights_kernel<<<(pack_threads + 255) / 256, 256, 0, stream>>>(
        Whh0, Wih1, Whh1, bih0, bhh0, bih1, bhh1, wpack, bsum);

    lstm_decode_kernel<<<NWG, WAVES * 32, 0, stream>>>(
        h, c, Wih0, fcw, fcb,
        wpack, wpack + (size_t)MATELEMS, wpack + (size_t)2 * MATELEMS,
        bsum, out);
}
